// FFFLayer_52012053955262
// MI455X (gfx1250) — matrix operand order repaired and checked by reading, not compiled, not run on hardware
//
#include <hip/hip_runtime.h>
#include <hip/hip_bf16.h>

typedef __attribute__((ext_vector_type(16))) __bf16 v16bf;
typedef __attribute__((ext_vector_type(8)))  __bf16 v8bf;
typedef __attribute__((ext_vector_type(8)))  float  v8f;
typedef __attribute__((ext_vector_type(4)))  float  f32x4;

#define IN_W   4096
#define OUT_W  4096
#define NWAVES 32
#define KSLICE (IN_W / NWAVES)   // 128 -> 4 WMMA k-chunks per wave
#define NCHUNK (KSLICE / 32)     // 4
#define YCHUNK 8                 // per-wave y chunks: 2048 cols / (32 lanes * 8)

// One workgroup = 32 waves (1024 threads), owns 16 samples.
// y accumulators live in (dynamic) LDS: sample s row = sh_y[s*4096 ..], each
// wave RMWs its private half. Each wave preloads a 128-wide x K-slice as WMMA
// A operands (HBM read once). Per tree level:
//   Phase 1: ALL 32 waves batch-issue the full w1-row gather (8 x b128), then
//            run 4 WMMAs (sched_barrier keeps loads ahead of the WMMAs);
//            partial logit diagonals combine via ds_add_f32.
//   Phase 2: threads 0..15 apply branchless sigmoid-form GELU + branch update.
//   Phase 3: batch-issue 8 w2 loads, then packed-bf16 FMA RMW of y in LDS.
extern "C" __global__ __launch_bounds__(1024, 1)
void fff_tree_kernel(const __bf16* __restrict__ x,
                     const __bf16* __restrict__ w1,
                     const __bf16* __restrict__ w2,
                     __bf16* __restrict__ out,
                     const int* __restrict__ depth_p)
{
    extern __shared__ __bf16 sh_y[];   // 16 * 4096 bf16 = 128 KB dynamic LDS

    __shared__ float sh_logit[16];   // f32 partial-sum accumulators (atomic)
    __shared__ float sh_g[16];       // gelu(logit) per sample
    __shared__ int   sh_w2node[16];  // node used for this level's w2 gather
    __shared__ int   sh_nodes[16];   // current tree node per sample

    const int tid   = threadIdx.x;
    const int lane  = tid & 31;
    const int wave  = tid >> 5;
    const int depth = *depth_p;

    if (tid < 16) { sh_nodes[tid] = 0; sh_logit[tid] = 0.0f; }

    const size_t base_row = (size_t)blockIdx.x * 16;
    const int m  = lane & 15;   // sample (A row) / w1 column (B col)
    const int hi = lane >> 4;   // half-wave selects K sub-range

    // ---- Preload this wave's x K-slice as WMMA A operands (once, from HBM).
    // A (16x32 bf16): hi=0 -> K={0..7,16..23}, hi=1 -> K={8..15,24..31}
    v16bf aK[NCHUNK];
    {
        const __bf16* xrow = x + (base_row + m) * IN_W + wave * KSLICE;
#pragma unroll
        for (int c = 0; c < NCHUNK; ++c) {
            union UA { f32x4 q[2]; v16bf v; } ua;
            ua.q[0] = *(const f32x4*)(xrow + c * 32 + hi * 8);
            ua.q[1] = *(const f32x4*)(xrow + c * 32 + 16 + hi * 8);
            aK[c] = ua.v;
        }
    }

    // y region: wave (2s+h) owns sample s cols [h*2048, h*2048+2048) in LDS;
    // lane owns 8 chunks of 8 bf16 (cols h*2048 + t*256 + lane*8 .. +7)
    const int ysample = wave >> 1;
    const int yhalf   = wave & 1;
    v8bf* yb = (v8bf*)(sh_y + ysample * OUT_W + yhalf * (OUT_W / 2));
    {
        v8bf zero;
#pragma unroll
        for (int j = 0; j < 8; ++j) zero[j] = (__bf16)0.0f;
#pragma unroll
        for (int t = 0; t < YCHUNK; ++t)
            yb[t * 32 + lane] = zero;
    }

    __syncthreads();

    int node_m = sh_nodes[m];        // register-tracked node for WMMA column m

    for (int lev = 0; lev < depth; ++lev) {
        // ---- Phase 1: all waves, WMMA partial logits over K slice ----
        {
            // B (32x16 bf16): lane column N=m, K = k0 + hi*8 + {0..7,16..23}
            const __bf16* wrow = w1 + (size_t)node_m * IN_W
                                    + wave * KSLICE + hi * 8;

            union UB { f32x4 q[2]; v16bf v; };
            UB b[NCHUNK];
#pragma unroll
            for (int c = 0; c < NCHUNK; ++c) {
                b[c].q[0] = *(const f32x4*)(wrow + c * 32);
                b[c].q[1] = *(const f32x4*)(wrow + c * 32 + 16);
            }
            __builtin_amdgcn_sched_barrier(0);   // keep all gathers ahead

            v8f acc = {};
#pragma unroll
            for (int c = 0; c < NCHUNK; ++c)
                acc = __builtin_amdgcn_wmma_f32_16x16x32_bf16(
                    false, aK[c], false, b[c].v, (short)0, acc, false, false);

            // Diagonal lives at: i<8 -> lane i, acc[i]; i>=8 -> lane i+16, acc[i-8]
            float d = acc[0];
#pragma unroll
            for (int rr = 1; rr < 8; ++rr)
                if ((lane & 7) == rr) d = acc[rr];
            if (lane < 8 || lane >= 24)
                atomicAdd(&sh_logit[(lane < 8) ? lane : (lane - 16)], d);
        }
        __syncthreads();

        // ---- Phase 2: branchless GELU + branch update (threads 0..15) ----
        // gelu_tanh(x) = x * sigmoid(2*beta*(x + 0.044715 x^3)); robust at both
        // tails (exp2->inf => rcp->0 => g->0 ; exp2->0 => g->x).
        if (tid < 16) {
            const float logit = sh_logit[tid];
            sh_logit[tid] = 0.0f;                       // re-arm for next level
            const float kBeta  = 0.7978845608028654f;   // sqrt(2/pi)
            const float kLog2e = 1.4426950408889634f;
            const float z  = kBeta * (logit + 0.044715f * logit * logit * logit);
            const float e  = __builtin_amdgcn_exp2f(-2.0f * kLog2e * z);
            const float g  = logit * __builtin_amdgcn_rcpf(1.0f + e);
            const int node  = sh_nodes[tid];
            sh_g[tid]       = g;
            sh_w2node[tid]  = node;                     // pre-branch node for w2
            sh_nodes[tid]   = 2 * node + 1 + (logit > 0.0f ? 1 : 0);
        }
        __syncthreads();

        // ---- Phase 3: y[s] += g[s] * w2[node_s] (packed bf16, y in LDS) ----
        {
            const int node = sh_w2node[ysample];
            const v8bf* w2row = (const v8bf*)(w2 + (size_t)node * OUT_W
                                                 + yhalf * (OUT_W / 2));
            v8bf wv[YCHUNK];
#pragma unroll
            for (int t = 0; t < YCHUNK; ++t)
                wv[t] = w2row[t * 32 + lane];           // coalesced 16B per lane
            __builtin_amdgcn_sched_barrier(0);          // keep all loads ahead

            const __bf16 gb = (__bf16)sh_g[ysample];
            v8bf gv;
#pragma unroll
            for (int j = 0; j < 8; ++j) gv[j] = gb;

            node_m = sh_nodes[m];        // prefetch next level's node (post-B2)
#pragma unroll
            for (int t = 0; t < YCHUNK; ++t) {
                v8bf yv = yb[t * 32 + lane];
                yv += wv[t] * gv;                       // v_pk_fma_bf16
                yb[t * 32 + lane] = yv;
            }
        }
        // No barrier needed here: sh_logit was re-armed before the barrier
        // above, and sh_g/sh_w2node/sh_nodes are only rewritten after the next
        // level's __syncthreads().
    }

    // ---- Final store: LDS -> global, coalesced b128 ----
    v8bf* orow = (v8bf*)(out + (base_row + ysample) * OUT_W
                             + yhalf * (OUT_W / 2));
#pragma unroll
    for (int t = 0; t < YCHUNK; ++t)
        orow[t * 32 + lane] = yb[t * 32 + lane];
}

extern "C" void kernel_launch(void* const* d_in, const int* in_sizes, int n_in,
                              void* d_out, int out_size, void* d_ws, size_t ws_size,
                              hipStream_t stream) {
    const __bf16* x   = (const __bf16*)d_in[0];
    const __bf16* w1s = (const __bf16*)d_in[1];
    const __bf16* w2s = (const __bf16*)d_in[2];
    const int* depth  = (const int*)d_in[3];
    __bf16* out = (__bf16*)d_out;

    const int Bn = in_sizes[0] / IN_W;   // tokens
    dim3 grid(Bn / 16);
    dim3 block(1024);
    const size_t dyn_lds = (size_t)16 * OUT_W * sizeof(__bf16);  // 128 KB
    fff_tree_kernel<<<grid, block, dyn_lds, stream>>>(x, w1s, w2s, out, depth);
}
